// RPN_67267777790298
// MI455X (gfx1250) — matrix operand order repaired and checked by reading, not compiled, not run on hardware
//
#include <hip/hip_runtime.h>
#include <hip/hip_bf16.h>
#include <math.h>

// ---------------- types ----------------
typedef __attribute__((ext_vector_type(16))) __bf16    v16bf;
typedef __attribute__((ext_vector_type(8)))  float     v8f;
typedef __attribute__((ext_vector_type(4)))  unsigned  u32x4;

// ---------------- problem constants ----------------
#define B_IMG   16
#define C_IN    512
#define H_FM    38
#define W_FM    50
#define M_IMG   (H_FM * W_FM)      // 1900 spatial positions / image
#define N_ANCH  (M_IMG * 9)        // 17100 anchors / image
#define K_CONV  (C_IN * 9)         // 4608, ordered k = tap*512 + ci (tap-major)
#define N_OUT   512
#define K_HEAD  512
#define N_HEAD  64                 // 18 cls + 36 reg + 10 pad
#define POST_N  300
#define IMG_H_F 608.0f
#define IMG_W_F 800.0f
#define MIN_SZ  16.0f
#define NMS_TH  0.7f

// conv GEMM tiling: 64 x 128 block tile, BK=32, 8 waves each owning 32x32
#define BM 64
#define BN 128
#define BK 32
#define MT ((M_IMG + BM - 1) / BM)   // 30 M-tiles per image
#define LPAD 20                      // LDS row pitch in dwords (80B, 16B-aligned, bank-skewed)
// heads GEMM tiling: 64 x 64
#define HBN 64

// ---------------- bf16 helpers ----------------
static __device__ __forceinline__ unsigned short f2bf_bits(float f) {
  union { float f; unsigned u; } a; a.f = f;
  unsigned r = a.u + 0x7FFFu + ((a.u >> 16) & 1u);   // RNE
  return (unsigned short)(r >> 16);
}

union FragB16 { v16bf v; unsigned u[8]; u32x4 q[2]; };

// ---------------- packing kernels ----------------
// x [B][C][HW] f32 -> xt [B][HW][C] bf16 (channels-last). grid=(HW*C/256, B)
__global__ void k_cvt_x(const float* __restrict__ x, unsigned short* __restrict__ xt) {
  int idx = blockIdx.x * blockDim.x + threadIdx.x;   // pos*512 + ci
  int b = blockIdx.y;
  int pos = idx >> 9, ci = idx & 511;
  xt[((size_t)b * M_IMG << 9) + idx] = f2bf_bits(x[((size_t)(b * C_IN + ci)) * M_IMG + pos]);
}
// zero the 512-entry pad rows appended to xt and sht (zero-row redirect target)
__global__ void k_zero_pads(unsigned short* __restrict__ xt_z,
                            unsigned short* __restrict__ sht_z) {
  int t = threadIdx.x;            // one block of 256
  xt_z[t] = 0;  xt_z[t + 256] = 0;
  sht_z[t] = 0; sht_z[t + 256] = 0;
}
// share_w [N][Ci][9] -> BwF1 fragment-major: dword (kpair) index
//   dst_dword = (chunk*512 + n)*16 + kp   with k = tap*512+ci, chunk=k>>5, kp=(k>>1)&15
__global__ void k_pack_w1(const float* __restrict__ w, unsigned short* __restrict__ BwF1) {
  int i = blockIdx.x * blockDim.x + threadIdx.x;
  if (i >= K_CONV * N_OUT) return;
  int k = i >> 9, n = i & 511;            // i = k*512 + n
  int tap = k >> 9, ci = k & 511;         // k = tap*512 + ci
  float v = w[(n * C_IN + ci) * 9 + tap];
  size_t dst = ((((size_t)(k >> 5)) * N_OUT + n) * 16 + ((k >> 1) & 15)) * 2 + (k & 1);
  BwF1[dst] = f2bf_bits(v);
}
// cls_w[18][512] + reg_w[36][512] -> BwF2 fragment-major [chunk][64][16 kpairs][2]
__global__ void k_pack_w2(const float* __restrict__ cw, const float* __restrict__ rw,
                          unsigned short* __restrict__ BwF2) {
  int i = blockIdx.x * blockDim.x + threadIdx.x;
  if (i >= K_HEAD * N_HEAD) return;
  int k = i >> 6, n = i & 63;
  float v = 0.0f;
  if (n < 18)      v = cw[n * K_HEAD + k];
  else if (n < 54) v = rw[(n - 18) * K_HEAD + k];
  size_t dst = ((((size_t)(k >> 5)) * N_HEAD + n) * 16 + ((k >> 1) & 15)) * 2 + (k & 1);
  BwF2[dst] = f2bf_bits(v);
}

// ---------------- 3x3 conv: implicit GEMM, bf16 WMMA ----------------
// grid = (MT, N_OUT/BN, B_IMG), block = 256 (8 waves, each 32x32 out)
__global__ __launch_bounds__(256)
void k_conv3x3_wmma(const unsigned short* __restrict__ xt,   // [B][HW+1pad][512] bf16
                    const unsigned short* __restrict__ BwF,  // fragment-major weights
                    const float* __restrict__ bias,
                    unsigned short* __restrict__ sht /* [B][HW][512] bf16 */) {
  __shared__ __attribute__((aligned(16))) unsigned As32[BM][LPAD];  // row-major K-pairs
  __shared__ __attribute__((aligned(16))) unsigned Bs32[BN][LPAD];  // col-major K-pairs
  const int b   = blockIdx.z;
  const int m0  = blockIdx.x * BM;
  const int n0  = blockIdx.y * BN;
  const int tid = threadIdx.x;
  const int lane = tid & 31, wave = tid >> 5;
  const int mw = wave & 1;        // M half (rows mw*32)
  const int nw = wave >> 1;       // N quarter (cols nw*32)
  const int lm = lane & 15, hi = lane >> 4;

  // per-thread A-staging invariants (one b128 per K-step)
  const int tAm = tid >> 2;               // row in tile 0..63
  const int tAd = (tid & 3) * 4;          // dword offset in row
  const int gmA = m0 + tAm;
  const int yA = gmA / W_FM, xA = gmA - yA * W_FM;
  const bool mOK = gmA < M_IMG;
  const unsigned short* const zrow = xt + ((size_t)B_IMG * M_IMG << 9);  // 512 zeros

  // per-thread B-staging invariants: one column x 8 k-pairs (2 b128) per K-step
  const int tBc  = tid >> 1;              // column 0..127
  const int tBk  = (tid & 1) * 8;         // k-pair start 0 or 8
  const size_t bStep = (size_t)N_OUT * 16 * 2;   // elements per K-chunk of weights

  v8f acc[2][2] = {};

  for (int tap = 0; tap < 9; ++tap) {
    const int ky = (tap >= 6) ? 2 : (tap >= 3 ? 1 : 0);
    const int kx = tap - ky * 3;
    const int iy = yA + ky - 1, ix = xA + kx - 1;
    const bool ok = mOK && (unsigned)iy < (unsigned)H_FM && (unsigned)ix < (unsigned)W_FM;
    // per-tap A pointer; invalid threads walk the zero row (exactly 512 elems/tap)
    const unsigned short* pA =
        (ok ? xt + (((size_t)b * M_IMG + iy * W_FM + ix) << 9) : zrow) + tAd * 2;
    // per-tap B pointer into fragment-major weights
    const unsigned short* pB =
        BwF + (((size_t)tap * 16 * N_OUT + n0 + tBc) * 16 + tBk) * 2;

    for (int cc = 0; cc < 16; ++cc) {
      // ---- stage A: one contiguous b128 per thread
      u32x4 a = *(const u32x4*)pA;
      *(u32x4*)&As32[tAm][tAd] = a;
      // ---- stage B: two contiguous b128 per thread (one column, 8 k-pairs)
      u32x4 b0 = *(const u32x4*)pB;
      u32x4 b1 = *(const u32x4*)(pB + 8);
      *(u32x4*)&Bs32[tBc][tBk]     = b0;
      *(u32x4*)&Bs32[tBc][tBk + 4] = b1;
      // ---- prefetch next chunk's source lines (global_prefetch_b8)
      __builtin_prefetch(pA + 32, 0, 1);
      __builtin_prefetch(pB + bStep, 0, 1);
      pA += 32;          // next 32 channels
      pB += bStep;       // next K-chunk
      __syncthreads();

      // ---- fragments: each is exactly two aligned ds_load_b128
      FragB16 fa[2], fb[2];
      #pragma unroll
      for (int i = 0; i < 2; ++i) {
        const int row = mw * 32 + i * 16 + lm;
        fa[i].q[0] = *(const u32x4*)&As32[row][hi * 8];
        fa[i].q[1] = *(const u32x4*)&As32[row][hi * 8 + 4];
      }
      #pragma unroll
      for (int j = 0; j < 2; ++j) {
        const int col = nw * 32 + j * 16 + lm;
        fb[j].q[0] = *(const u32x4*)&Bs32[col][hi * 8];
        fb[j].q[1] = *(const u32x4*)&Bs32[col][hi * 8 + 4];
      }
      #pragma unroll
      for (int i = 0; i < 2; ++i)
        #pragma unroll
        for (int j = 0; j < 2; ++j)
          acc[i][j] = __builtin_amdgcn_wmma_f32_16x16x32_bf16(
              false, fa[i].v, false, fb[j].v, (short)0, acc[i][j], false, false);
      __syncthreads();
    }
  }

  // ---- epilogue: bias + ReLU, channels-last bf16 (coalesced in n)
  #pragma unroll
  for (int i = 0; i < 2; ++i) {
    #pragma unroll
    for (int r = 0; r < 8; ++r) {
      const int gm = m0 + mw * 32 + i * 16 + r + hi * 8;
      if (gm < M_IMG) {
        #pragma unroll
        for (int j = 0; j < 2; ++j) {
          const int n = n0 + nw * 32 + j * 16 + lm;
          float v = acc[i][j][r] + bias[n];
          v = v > 0.0f ? v : 0.0f;
          sht[(((size_t)b * M_IMG + gm) << 9) + n] = f2bf_bits(v);
        }
      }
    }
  }
}

// ---------------- fused 1x1 heads: 64 x 64 GEMM, K=512 ----------------
// grid = (MT, 1, B_IMG), block = 256 (8 waves: 4 M-strips x 2 N-halves)
__global__ __launch_bounds__(256)
void k_heads_wmma(const unsigned short* __restrict__ sht,   // [B][HW+1pad][512] bf16
                  const unsigned short* __restrict__ BwF2,  // fragment-major weights
                  const float* __restrict__ cls_b, const float* __restrict__ reg_b,
                  float* __restrict__ cls_raw /* [B][1900][18] */,
                  float* __restrict__ loc     /* [B][17100][4] */) {
  __shared__ __attribute__((aligned(16))) unsigned As32[BM][LPAD];
  __shared__ __attribute__((aligned(16))) unsigned Bs32[HBN][LPAD];
  const int b   = blockIdx.z;
  const int m0  = blockIdx.x * BM;
  const int tid = threadIdx.x;
  const int lane = tid & 31, wave = tid >> 5;
  const int mi = wave & 3, ng = wave >> 2;
  const int lm = lane & 15, hi = lane >> 4;

  const int tAm = tid >> 2;
  const int tAd = (tid & 3) * 4;
  const int gmA = m0 + tAm;
  const unsigned short* const zrow = sht + ((size_t)B_IMG * M_IMG << 9);
  const unsigned short* pA =
      (gmA < M_IMG ? sht + (((size_t)b * M_IMG + gmA) << 9) : zrow) + tAd * 2;
  // B: 64 cols x 16 k-pairs; first 128 threads: one column half (8 k-pairs) each
  const int tBc = (tid & 127) >> 1;
  const int tBk = (tid & 1) * 8;
  const size_t bStep = (size_t)N_HEAD * 16 * 2;
  const unsigned short* pB = BwF2 + (((size_t)tBc) * 16 + tBk) * 2;

  v8f acc0 = {}, acc1 = {};

  for (int cc = 0; cc < 16; ++cc) {
    u32x4 a = *(const u32x4*)pA;
    *(u32x4*)&As32[tAm][tAd] = a;
    if (tid < 128) {
      u32x4 b0 = *(const u32x4*)pB;
      u32x4 b1 = *(const u32x4*)(pB + 8);
      *(u32x4*)&Bs32[tBc][tBk]     = b0;
      *(u32x4*)&Bs32[tBc][tBk + 4] = b1;
    }
    pA += 32;
    pB += bStep;
    __syncthreads();

    FragB16 fa, fb0, fb1;
    {
      const int row = mi * 16 + lm;
      fa.q[0] = *(const u32x4*)&As32[row][hi * 8];
      fa.q[1] = *(const u32x4*)&As32[row][hi * 8 + 4];
      const int c0 = ng * 32 + lm;
      fb0.q[0] = *(const u32x4*)&Bs32[c0][hi * 8];
      fb0.q[1] = *(const u32x4*)&Bs32[c0][hi * 8 + 4];
      fb1.q[0] = *(const u32x4*)&Bs32[c0 + 16][hi * 8];
      fb1.q[1] = *(const u32x4*)&Bs32[c0 + 16][hi * 8 + 4];
    }
    acc0 = __builtin_amdgcn_wmma_f32_16x16x32_bf16(false, fa.v, false, fb0.v,
                                                   (short)0, acc0, false, false);
    acc1 = __builtin_amdgcn_wmma_f32_16x16x32_bf16(false, fa.v, false, fb1.v,
                                                   (short)0, acc1, false, false);
    __syncthreads();
  }

  #pragma unroll
  for (int r = 0; r < 8; ++r) {
    const int gm = m0 + mi * 16 + r + hi * 8;
    if (gm < M_IMG) {
      #pragma unroll
      for (int j = 0; j < 2; ++j) {
        const int n = ng * 32 + j * 16 + lm;
        const float v = (j == 0 ? acc0[r] : acc1[r]);
        if (n < 18) {
          cls_raw[((size_t)b * M_IMG + gm) * 18 + n] = v + cls_b[n];
        } else if (n < 54) {
          const int c = n - 18;
          loc[((size_t)b * N_ANCH + (size_t)gm * 9 + (c >> 2)) * 4 + (c & 3)] = v + reg_b[c];
        }
      }
    }
  }
}

// ---------------- softmax(18) + anchor decode + clip + min-size -----------
__global__ void k_decode(const float* __restrict__ cls_raw, const float* __restrict__ loc,
                         float* __restrict__ boxes, float* __restrict__ scores) {
  int idx = blockIdx.x * blockDim.x + threadIdx.x;
  if (idx >= B_IMG * M_IMG) return;
  int b = idx / M_IMG, m = idx - b * M_IMG;
  const float* c = cls_raw + (size_t)(b * M_IMG + m) * 18;
  float mx = c[0];
  #pragma unroll
  for (int i = 1; i < 18; ++i) mx = fmaxf(mx, c[i]);
  float e[18]; float sum = 0.0f;
  #pragma unroll
  for (int i = 0; i < 18; ++i) { e[i] = __expf(c[i] - mx); sum += e[i]; }
  float inv = 1.0f / sum;
  int y = m / W_FM, x = m - y * W_FM;
  const float ratios[3] = {0.5f, 1.0f, 2.0f};
  const float scales[3] = {8.0f, 16.0f, 32.0f};
  #pragma unroll
  for (int a = 0; a < 9; ++a) {
    int ri = a / 3, si = a - ri * 3;
    float hh = 16.0f * scales[si] * __fsqrt_rn(ratios[ri]);
    float ww = 16.0f * scales[si] * __fsqrt_rn(1.0f / ratios[ri]);
    float sy = y * 16.0f, sx = x * 16.0f;
    float ay1 = sy + 8.0f - 0.5f * hh, ax1 = sx + 8.0f - 0.5f * ww;
    float ay2 = sy + 8.0f + 0.5f * hh, ax2 = sx + 8.0f + 0.5f * ww;
    float ah = ay2 - ay1, aw = ax2 - ax1;
    float acy = ay1 + 0.5f * ah, acx = ax1 + 0.5f * aw;
    int n = m * 9 + a;
    const float* l = loc + ((size_t)b * N_ANCH + n) * 4;
    float cy = l[0] * ah + acy, cx = l[1] * aw + acx;
    float bh = __expf(l[2]) * ah, bw = __expf(l[3]) * aw;
    float y1 = fminf(fmaxf(cy - 0.5f * bh, 0.0f), IMG_H_F);
    float x1 = fminf(fmaxf(cx - 0.5f * bw, 0.0f), IMG_W_F);
    float y2 = fminf(fmaxf(cy + 0.5f * bh, 0.0f), IMG_H_F);
    float x2 = fminf(fmaxf(cx + 0.5f * bw, 0.0f), IMG_W_F);
    bool keep = ((y2 - y1) >= MIN_SZ) && ((x2 - x1) >= MIN_SZ);
    float* bo = boxes + ((size_t)b * N_ANCH + n) * 4;
    bo[0] = y1; bo[1] = x1; bo[2] = y2; bo[3] = x2;
    scores[(size_t)b * N_ANCH + n] = keep ? (e[2 * a + 1] * inv) : -__builtin_inff();
  }
}

// ---------------- greedy NMS: one workgroup per image ---------------------
#define NMS_T 1024
__global__ __launch_bounds__(NMS_T)
void k_nms(const float* __restrict__ boxes, const float* __restrict__ scores,
           float* __restrict__ out) {
  __shared__ unsigned char valid[N_ANCH];
  __shared__ float rv[NMS_T];
  __shared__ int   ri[NMS_T];
  __shared__ float bb[4];
  __shared__ int   bsel;
  const int b = blockIdx.x, t = threadIdx.x;
  const float* bx = boxes  + (size_t)b * N_ANCH * 4;
  const float* sc = scores + (size_t)b * N_ANCH;
  float* o = out + (size_t)b * POST_N * 5;
  for (int e = t; e < N_ANCH; e += NMS_T) valid[e] = (sc[e] > -1e30f) ? 1 : 0;
  __syncthreads();

  for (int it = 0; it < POST_N; ++it) {
    float best = -__builtin_inff(); int bi = 0x7fffffff;
    for (int e = t; e < N_ANCH; e += NMS_T) {
      if (valid[e]) {
        float s = sc[e];
        if (s > best || (s == best && e < bi)) { best = s; bi = e; }
      }
    }
    rv[t] = best; ri[t] = bi;
    __syncthreads();
    for (int s = NMS_T / 2; s > 0; s >>= 1) {
      if (t < s) {
        float ov = rv[t + s]; int oi = ri[t + s];
        if (ov > rv[t] || (ov == rv[t] && oi < ri[t])) { rv[t] = ov; ri[t] = oi; }
      }
      __syncthreads();
    }
    if (t == 0) {
      int i = ri[0]; float s = rv[0];
      if (i != 0x7fffffff && s > -1e30f) {
        bsel = i;
        bb[0] = bx[i * 4 + 0]; bb[1] = bx[i * 4 + 1];
        bb[2] = bx[i * 4 + 2]; bb[3] = bx[i * 4 + 3];
        valid[i] = 0;
        o[it * 5 + 0] = bb[0]; o[it * 5 + 1] = bb[1];
        o[it * 5 + 2] = bb[2]; o[it * 5 + 3] = bb[3];
        o[it * 5 + 4] = s;
      } else {
        bsel = -1;
        o[it * 5 + 0] = 0.0f; o[it * 5 + 1] = 0.0f; o[it * 5 + 2] = 0.0f;
        o[it * 5 + 3] = 0.0f; o[it * 5 + 4] = 0.0f;
      }
    }
    __syncthreads();
    if (bsel >= 0) {
      float by1 = bb[0], bx1 = bb[1], by2 = bb[2], bx2 = bb[3];
      float barea = fmaxf(by2 - by1, 0.0f) * fmaxf(bx2 - bx1, 0.0f);
      for (int e = t; e < N_ANCH; e += NMS_T) {
        if (valid[e]) {
          float y1 = bx[e * 4 + 0], x1 = bx[e * 4 + 1];
          float y2 = bx[e * 4 + 2], x2 = bx[e * 4 + 3];
          float iy1 = fmaxf(by1, y1), ix1 = fmaxf(bx1, x1);
          float iy2 = fminf(by2, y2), ix2 = fminf(bx2, x2);
          float inter = fmaxf(iy2 - iy1, 0.0f) * fmaxf(ix2 - ix1, 0.0f);
          float area  = fmaxf(y2 - y1, 0.0f) * fmaxf(x2 - x1, 0.0f);
          float iou = inter / (area + barea - inter + 1e-9f);
          if (iou > NMS_TH) valid[e] = 0;
        }
      }
    }
    __syncthreads();
  }
}

// ---------------- launch ----------------
extern "C" void kernel_launch(void* const* d_in, const int* in_sizes, int n_in,
                              void* d_out, int out_size, void* d_ws, size_t ws_size,
                              hipStream_t stream) {
  (void)in_sizes; (void)n_in; (void)out_size; (void)ws_size;
  const float* x       = (const float*)d_in[0];
  const float* share_w = (const float*)d_in[1];
  const float* share_b = (const float*)d_in[2];
  const float* cls_w   = (const float*)d_in[3];
  const float* cls_b   = (const float*)d_in[4];
  const float* reg_w   = (const float*)d_in[5];
  const float* reg_b   = (const float*)d_in[6];
  float* out = (float*)d_out;

  char* p = (char*)d_ws;
  auto take = [&](size_t bytes) { void* r = (void*)p; p += (bytes + 255) & ~(size_t)255; return r; };
  // +512 bf16 pad row on xt and sht for the zero-row redirect
  unsigned short* xt   = (unsigned short*)take(((size_t)B_IMG * M_IMG * C_IN + 512) * 2);
  unsigned short* BwF1 = (unsigned short*)take((size_t)K_CONV * N_OUT * 2);
  unsigned short* BwF2 = (unsigned short*)take((size_t)K_HEAD * N_HEAD * 2);
  unsigned short* sht  = (unsigned short*)take(((size_t)B_IMG * M_IMG * N_OUT + 512) * 2);
  float* cls_raw       = (float*)take((size_t)B_IMG * M_IMG * 18 * 4);
  float* loc           = (float*)take((size_t)B_IMG * N_ANCH * 4 * 4);
  float* boxes         = (float*)take((size_t)B_IMG * N_ANCH * 4 * 4);
  float* scores        = (float*)take((size_t)B_IMG * N_ANCH * 4);

  k_cvt_x<<<dim3((M_IMG * C_IN) / 256, B_IMG), 256, 0, stream>>>(x, xt);
  k_zero_pads<<<1, 256, 0, stream>>>(xt + ((size_t)B_IMG * M_IMG << 9),
                                     sht + ((size_t)B_IMG * M_IMG << 9));
  k_pack_w1<<<(K_CONV * N_OUT + 255) / 256, 256, 0, stream>>>(share_w, BwF1);
  k_pack_w2<<<(K_HEAD * N_HEAD + 255) / 256, 256, 0, stream>>>(cls_w, reg_w, BwF2);

  k_conv3x3_wmma<<<dim3(MT, N_OUT / BN, B_IMG), 256, 0, stream>>>(xt, BwF1, share_b, sht);
  k_heads_wmma<<<dim3(MT, 1, B_IMG), 256, 0, stream>>>(sht, BwF2, cls_b, reg_b, cls_raw, loc);

  {
    int n = B_IMG * M_IMG;
    k_decode<<<(n + 255) / 256, 256, 0, stream>>>(cls_raw, loc, boxes, scores);
  }
  k_nms<<<B_IMG, NMS_T, 0, stream>>>(boxes, scores, out);
}
